// MLP_78254304133739
// MI455X (gfx1250) — matrix operand order repaired and checked by reading, not compiled, not run on hardware
//
#include <hip/hip_runtime.h>
#include <hip/hip_bf16.h>

typedef __attribute__((ext_vector_type(16))) _Float16 v16h;
typedef __attribute__((ext_vector_type(8)))  _Float16 v8h;
typedef __attribute__((ext_vector_type(8)))  float    v8f;

#define T_LEN  365
#define K1P    384
#define N1     512
#define N2     256
#define N3     128
#define KC     160
#define NC1    64
#define NC2    32
#define ROWS   32            // rows per (single-wave) workgroup

// ---------------- weight transpose + f32->f16 convert -----------------------
__global__ void convert_wt(const float* __restrict__ W, _Float16* __restrict__ Wt,
                           int Kin, int Kpad, int N) {
    int idx = blockIdx.x * blockDim.x + threadIdx.x;
    int total = N * Kpad;
    if (idx >= total) return;
    int n = idx / Kpad;
    int k = idx - n * Kpad;
    Wt[idx] = (k < Kin) ? (_Float16)W[k * N + n] : (_Float16)0.0f;
}

// ------------------------- WMMA fragment helpers ----------------------------
// A fragment (16x32 f16): lanes 0-15 row M=lane hold K k0..k0+15;
// lanes 16-31 row M=lane-16 hold K k0+16..k0+31.
__device__ __forceinline__ v16h load_a_lds(const _Float16* base, int stride,
                                           int k0, int lane) {
    const int m  = lane & 15;
    const int kb = k0 + ((lane & 16) ? 16 : 0);
    const _Float16* p = base + m * stride + kb;
    return *(const v16h*)(p);
}

// B fragment (32x16 f16): column N=lane&15; lanes 0-15 hold K k0..k0+15,
// lanes 16-31 hold K k0+16..k0+31.  Weights stored [N][K] f16 -> one
// contiguous 32-byte load per lane.
__device__ __forceinline__ v16h load_b_glb(const _Float16* __restrict__ wt, int K,
                                           int nt, int k0, int lane) {
    const int n  = nt * 16 + (lane & 15);
    const int kb = k0 + ((lane & 16) ? 16 : 0);
    return *(const v16h*)(wt + n * K + kb);
}

// D tile (16x16 f32, 8 VGPRs): row = v + (lane>=16 ? 8:0), col = lane&15.
__device__ __forceinline__ void store_d_relu(_Float16* out, int stride, int nt,
                                             v8f c, const float* __restrict__ bias,
                                             int lane) {
    const int col = nt * 16 + (lane & 15);
    const float b = bias[col];
    const int r0 = (lane & 16) ? 8 : 0;
#pragma unroll
    for (int v = 0; v < 8; ++v) {
        float val = fmaxf(c[v] + b, 0.0f);
        out[(r0 + v) * stride + col] = (_Float16)val;
    }
}

// GEMM layer for a 32-row tile: two independent 16-row accumulator chains
// share every B fragment (halves L2 weight traffic, doubles WMMA ILP).
__device__ __forceinline__ void gemm2(const _Float16* A, int K,
                                      const _Float16* __restrict__ Wt,
                                      const float* __restrict__ bias,
                                      _Float16* Out, int outStride,
                                      int Ntiles, int lane) {
    for (int nt = 0; nt < Ntiles; ++nt) {
        v8f c0 = {};
        v8f c1 = {};
        for (int k0 = 0; k0 < K; k0 += 32) {
            v16h b  = load_b_glb(Wt, K, nt, k0, lane);
            v16h a0 = load_a_lds(A,          K, k0, lane);
            v16h a1 = load_a_lds(A + 16 * K, K, k0, lane);
            c0 = __builtin_amdgcn_wmma_f32_16x16x32_f16(
                     false, a0, false, b, (short)0, c0, false, false);
            c1 = __builtin_amdgcn_wmma_f32_16x16x32_f16(
                     false, a1, false, b, (short)0, c1, false, false);
        }
        store_d_relu(Out,                   outStride, nt, c0, bias, lane);
        store_d_relu(Out + 16 * outStride,  outStride, nt, c1, bias, lane);
    }
}

// ----------------------------- fused forward --------------------------------
__global__ void __launch_bounds__(32, 1)
fused_mlp(const float* __restrict__ x,
          const _Float16* __restrict__ w1t, const float* __restrict__ b1,
          const _Float16* __restrict__ w2t, const float* __restrict__ b2,
          const _Float16* __restrict__ w3t, const float* __restrict__ b3,
          const float* __restrict__ Ws, const float* __restrict__ bs,
          const _Float16* __restrict__ wc1t, const float* __restrict__ bc1,
          const _Float16* __restrict__ wc2t, const float* __restrict__ bc2,
          const float* __restrict__ Wc3, const float* __restrict__ bc3,
          float* __restrict__ out) {
    const int lane = threadIdx.x;            // single wave32 per block
    const int row0 = blockIdx.x * ROWS;

    // bufA = 32x384 halves (24 KB), bufB = 32x512 halves (32 KB)
    __shared__ __align__(32) _Float16 smem[ROWS * K1P + ROWS * N1];
    __shared__ float stats[ROWS][6];

    _Float16* bufA = smem;
    _Float16* bufB = smem + ROWS * K1P;

    // ---- 1) load x tile -> LDS f16 [32 x 384] (zero-padded K) ----
    for (int r = 0; r < ROWS; ++r) {
        const float* xr = x + (size_t)(row0 + r) * T_LEN;
        for (int k = lane; k < K1P; k += 32)
            bufA[r * K1P + k] = (k < T_LEN) ? (_Float16)xr[k] : (_Float16)0.0f;
    }

    // ---- 2) f32 stats, one lane per row (rows hot in L2 after pass 1) ----
    {
        const int r = lane;
        const float* xr = x + (size_t)(row0 + r) * T_LEN;
        float sum = 0.0f, mn = 3.4e38f, mx = -3.4e38f;
        for (int j = 0; j < T_LEN; ++j) {
            float v = xr[j];
            sum += v; mn = fminf(mn, v); mx = fmaxf(mx, v);
        }
        const float mean = sum / (float)T_LEN;
        float s2 = 0.0f, s3 = 0.0f, s4 = 0.0f;
        for (int j = 0; j < T_LEN; ++j) {
            float cv = xr[j] - mean;
            float c2 = cv * cv;
            s2 += c2; s3 += c2 * cv; s4 += c2 * c2;
        }
        float var = s2 / (float)(T_LEN - 1);
        float sd  = sqrtf(var);
        float sk  = (s3 / (float)T_LEN) / (sd * var + 1e-8f);   // /std^3
        float ku  = (s4 / (float)T_LEN) / (var * var + 1e-8f);  // /std^4
        stats[r][0] = mean; stats[r][1] = sd;
        stats[r][2] = mn;   stats[r][3] = mx;
        stats[r][4] = sk;   stats[r][5] = ku;
    }

    // ---- 3) big GEMM chain (all ReLU) ----
    gemm2(bufA, K1P, w1t, b1, bufB, N1, N1 / 16, lane);  // h1 [32,512] -> bufB
    gemm2(bufB, N1,  w2t, b2, bufA, N2, N2 / 16, lane);  // h2 [32,256] -> bufA
    gemm2(bufA, N2,  w3t, b3, bufB, KC, N3 / 16, lane);  // h3 [32,128] -> bufB s=160

    // ---- 4) stat features (no ReLU) -> combined cols 128..159 ----
    {
        const int n = lane;                  // 0..31
        const float bsv = bs[n];
        for (int r = 0; r < ROWS; ++r) {
            float acc = bsv;
#pragma unroll
            for (int j = 0; j < 6; ++j) acc += stats[r][j] * Ws[j * 32 + n];
            bufB[r * KC + 128 + n] = (_Float16)acc;
        }
    }

    // ---- 5) head GEMMs ----
    gemm2(bufB, KC,  wc1t, bc1, bufA, NC1, NC1 / 16, lane);  // c1 [32,64] -> bufA
    gemm2(bufA, NC1, wc2t, bc2, bufB, NC2, NC2 / 16, lane);  // c2 [32,32] -> bufB

    // ---- 6) final dot + sigmoid, one lane per row ----
    {
        float acc = bc3[0];
#pragma unroll
        for (int k = 0; k < 32; ++k)
            acc += (float)bufB[lane * NC2 + k] * Wc3[k];
        float sig = 1.0f / (1.0f + __expf(-acc));
        out[row0 + lane] = sig * 4.0f + 6.0f;
    }
}

// ------------------------------- launcher -----------------------------------
extern "C" void kernel_launch(void* const* d_in, const int* in_sizes, int n_in,
                              void* d_out, int out_size, void* d_ws, size_t ws_size,
                              hipStream_t stream) {
    const float* x   = (const float*)d_in[0];
    const float* W1  = (const float*)d_in[1];
    const float* b1  = (const float*)d_in[2];
    const float* W2  = (const float*)d_in[3];
    const float* b2  = (const float*)d_in[4];
    const float* W3  = (const float*)d_in[5];
    const float* b3  = (const float*)d_in[6];
    const float* Ws  = (const float*)d_in[7];
    const float* bs  = (const float*)d_in[8];
    const float* Wc1 = (const float*)d_in[9];
    const float* bc1 = (const float*)d_in[10];
    const float* Wc2 = (const float*)d_in[11];
    const float* bc2 = (const float*)d_in[12];
    const float* Wc3 = (const float*)d_in[13];
    const float* bc3 = (const float*)d_in[14];

    _Float16* ws   = (_Float16*)d_ws;
    _Float16* w1t  = ws;                    // 512*384
    _Float16* w2t  = w1t  + 512 * 384;      // 256*512
    _Float16* w3t  = w2t  + 256 * 512;      // 128*256
    _Float16* wc1t = w3t  + 128 * 256;      // 64*160
    _Float16* wc2t = wc1t + 64 * 160;       // 32*64

    auto cvt = [&](const float* W, _Float16* Wt, int Kin, int Kpad, int N) {
        int total = N * Kpad;
        convert_wt<<<(total + 255) / 256, 256, 0, stream>>>(W, Wt, Kin, Kpad, N);
    };
    cvt(W1,  w1t,  365, K1P, N1);
    cvt(W2,  w2t,  N1,  N1,  N2);
    cvt(W3,  w3t,  N2,  N2,  N3);
    cvt(Wc1, wc1t, KC,  KC,  NC1);
    cvt(Wc2, wc2t, NC1, NC1, NC2);

    const int B = in_sizes[0] / T_LEN;      // 65536
    const int blocks = B / ROWS;            // 2048 single-wave blocks
    fused_mlp<<<blocks, 32, 0, stream>>>(
        x, w1t, b1, w2t, b2, w3t, b3, Ws, bs,
        wc1t, bc1, wc2t, bc2, Wc3, bc3, (float*)d_out);
}
